// ModulatedConv2d_60705067762162
// MI455X (gfx1250) — matrix operand order repaired and checked by reading, not compiled, not run on hardware
//
#include <hip/hip_runtime.h>
#include <hip/hip_bf16.h>
#include <stdint.h>

typedef __attribute__((ext_vector_type(16))) __bf16 v16bf;
typedef __attribute__((ext_vector_type(8)))  float  v8f;
typedef __attribute__((ext_vector_type(4)))  unsigned int u32x4;
typedef __attribute__((ext_vector_type(8)))  int i32x8;
typedef __attribute__((ext_vector_type(4)))  int i32x4;

#define B_    16
#define CIN   128
#define COUT  128
#define H_    128
#define W_    128
#define S_    512
#define EPS_  1e-8f

#if __has_builtin(__builtin_amdgcn_tensor_load_to_lds)
#define HAVE_TDM 1
#else
#define HAVE_TDM 0
#endif

// round-to-nearest-even float -> bf16 (avoids relying on __bf16 arithmetic)
__device__ __forceinline__ __bf16 f2bf(float f) {
    union { float f; uint32_t u; } x; x.f = f;
    uint32_t r = x.u + 0x7FFFu + ((x.u >> 16) & 1u);
    uint16_t h = (uint16_t)(r >> 16);
    return __builtin_bit_cast(__bf16, h);
}

// ---------------------------------------------------------------------------
// Kernel 1: s[b][ci] = style[b,:] . mod_w[ci,:] + mod_b[ci]   (16x128 tiny)
// ---------------------------------------------------------------------------
__global__ __launch_bounds__(256) void style_mod_kernel(
    const float* __restrict__ style, const float* __restrict__ mod_w,
    const float* __restrict__ mod_b, float* __restrict__ s)
{
    int t = blockIdx.x * 256 + threadIdx.x;
    if (t >= B_ * CIN) return;
    int b = t >> 7, ci = t & 127;
    const float* st = style + (size_t)b * S_;
    const float* mw = mod_w + (size_t)ci * S_;
    float acc = 0.f;
    for (int k = 0; k < S_; ++k) acc += st[k] * mw[k];
    s[t] = acc + mod_b[ci];
}

// ---------------------------------------------------------------------------
// Kernel 2: modulate, demodulate, pack to bf16 as wpack[b][kh][kw][co][ci]
// ---------------------------------------------------------------------------
__global__ __launch_bounds__(128) void pack_weights_kernel(
    const float* __restrict__ weight, const float* __restrict__ s,
    __bf16* __restrict__ wpack)
{
    __shared__ float red[128];
    int b  = blockIdx.x >> 7;
    int co = blockIdx.x & 127;
    int ci = threadIdx.x;

    float sv = s[b * CIN + ci];
    const float* wp = weight + ((size_t)co * CIN + ci) * 9;
    float wv[9];
    float sum = 0.f;
    #pragma unroll
    for (int tap = 0; tap < 9; ++tap) {
        float v = wp[tap] * sv;
        wv[tap] = v;
        sum += v * v;
    }
    red[ci] = sum;
    __syncthreads();
    for (int off = 64; off > 0; off >>= 1) {
        if (ci < off) red[ci] += red[ci + off];
        __syncthreads();
    }
    float d = rsqrtf(red[0] + EPS_);
    #pragma unroll
    for (int tap = 0; tap < 9; ++tap)
        wpack[((size_t)(b * 9 + tap) * COUT + co) * CIN + ci] = f2bf(wv[tap] * d);
}

// ---------------------------------------------------------------------------
// Kernel 3: conv as 9 shifted GEMMs via V_WMMA_F32_16X16X32_BF16.
// grid = B*H blocks (one image row each), 256 threads = 8 waves.
// Weight staging: TDM tensor_load_to_lds (1D, 98304 B) issued by wave 0,
// overlapped with VALU x-tile transpose+convert staging.
// ---------------------------------------------------------------------------
__global__ __launch_bounds__(256) void modconv_wmma_kernel(
    const float* __restrict__ x, const __bf16* __restrict__ wpack,
    float* __restrict__ out)
{
    extern __shared__ char smem[];
    __bf16* wlds = (__bf16*)smem;                        // [kw][co][ci] 3*128*128 @ LDS 0
    __bf16* xlds = (__bf16*)(smem + 3 * COUT * CIN * 2); // [col 0..129][ci 0..31]

    const int tid  = threadIdx.x;
    const int b    = blockIdx.x >> 7;
    const int y    = blockIdx.x & 127;
    const int wv   = tid >> 5;
    const int lane = tid & 31;
    const int g    = lane >> 4;   // lane group (K/M split per WMMA layout)
    const int n    = lane & 15;   // N column / M row within fragment
    const int co0  = wv * 16;

    v8f acc[8] = {};

    for (int kh = 0; kh < 3; ++kh) {
        __syncthreads();  // previous kh's LDS consumers done before overwrite

        // ---- stage 3 kw-slices of modulated weights for this kh (96 KB) ----
#if HAVE_TDM
        if (wv == 0) {
            uint64_t gaddr =
                (uint64_t)(uintptr_t)(wpack + (size_t)(b * 9 + kh * 3) * (COUT * CIN));
            // D# group0: count=1 | lds_addr=0 | global_addr | type=2 ("image")
            u32x4 g0;
            g0.x = 1u;
            g0.y = 0u;                                     // wlds at LDS offset 0
            g0.z = (unsigned)(gaddr & 0xFFFFFFFFu);
            g0.w = (unsigned)((gaddr >> 32) & 0x01FFFFFFu) | (2u << 30);
            // D# group1: data_size=3 (8B); 1-D: tensor_dim0=tile_dim0=12288 units
            i32x8 g1;
            g1[0] = 3 << 16;                 // wg_mask=0, data_size=8B
            g1[1] = (int)(12288u << 16);     // tensor_dim0[15:0] in [31:16]
            g1[2] = 0;                       // tensor_dim0 hi | tensor_dim1 lo
            g1[3] = (int)(12288u << 16);     // tile_dim0
            g1[4] = 0;                       // tile_dim1=0 (unused), tile_dim2=0
            g1[5] = 12288;                   // tensor_dim0_stride lo
            g1[6] = 0;
            g1[7] = 0;
            i32x4 z4 = {0, 0, 0, 0};
#if defined(__clang_major__) && (__clang_major__ >= 23)
            i32x8 z8 = {0, 0, 0, 0, 0, 0, 0, 0};
            __builtin_amdgcn_tensor_load_to_lds(g0, g1, z4, z4, z8, 0);
#else
            __builtin_amdgcn_tensor_load_to_lds(g0, g1, z4, z4, 0);
#endif
        }
#else
        {
            const uint4* src =
                (const uint4*)(wpack + (size_t)(b * 9 + kh * 3) * (COUT * CIN));
            uint4* dst = (uint4*)wlds;
            for (int i = tid; i < (3 * COUT * CIN) / 8; i += 256) dst[i] = src[i];
        }
#endif
        const int row = y + kh - 1;

        for (int cc = 0; cc < 4; ++cc) {   // C_in chunks of 32 (WMMA K)
            if (cc) __syncthreads();       // protect xlds from prev readers
            const int ci0 = cc * 32;

            // prefetch next ci-chunk's row segment into caches
            if (cc < 3 && (unsigned)row < (unsigned)H_) {
                int pci  = tid >> 3;            // 0..31
                int pcol = (tid & 7) << 4;      // 0,16,...,112
                __builtin_prefetch(
                    &x[(((size_t)b * CIN + ci0 + 32 + pci) * H_ + row) * W_ + pcol],
                    0, 1);
            }

            // stage x row-tile transposed: xlds[col][ci], col 0 <-> img col -1
            for (int idx = tid; idx < 130 * 32; idx += 256) {
                int ci  = idx / 130;
                int col = idx - ci * 130;
                int ic  = col - 1;
                float v = 0.f;
                if ((unsigned)row < (unsigned)H_ && (unsigned)ic < (unsigned)W_)
                    v = x[(((size_t)b * CIN + ci0 + ci) * H_ + row) * W_ + ic];
                xlds[col * 32 + ci] = f2bf(v);
            }
#if HAVE_TDM
            if (cc == 0 && wv == 0)
                __builtin_amdgcn_s_wait_tensorcnt((short)0);  // weights landed
#endif
            __syncthreads();

            #pragma unroll
            for (int kw = 0; kw < 3; ++kw) {
                // A fragment: 16x32 bf16. lane(m=n,g): K 16g..16g+15, two adjacent 16B runs
                const __bf16* arow =
                    wlds + kw * (COUT * CIN) + (co0 + n) * CIN + ci0 + g * 16;
                union { struct { uint4 lo; uint4 hi; } u; v16bf v; } au;
                au.u.lo = *(const uint4*)(arow);
                au.u.hi = *(const uint4*)(arow + 8);
                v16bf a = au.v;

                #pragma unroll
                for (int t = 0; t < 8; ++t) {
                    // B fragment: 32x16. lane n = pixel col, g picks K 0-15/16-31
                    const v16bf* bp =
                        (const v16bf*)(xlds + (t * 16 + n + kw) * 32 + g * 16);
                    acc[t] = __builtin_amdgcn_wmma_f32_16x16x32_bf16(
                        false, a, false, *bp, (short)0, acc[t], false, false);
                }
            }
        }
    }

    // D layout: lane(g,n), VGPR r -> (M = g*8 + r, N = n)
    float* outp = out + (((size_t)b * COUT + co0) * H_ + y) * W_;
    #pragma unroll
    for (int t = 0; t < 8; ++t)
        #pragma unroll
        for (int r = 0; r < 8; ++r)
            outp[(size_t)(g * 8 + r) * (H_ * W_) + t * 16 + n] = acc[t][r];
}

// ---------------------------------------------------------------------------
extern "C" void kernel_launch(void* const* d_in, const int* in_sizes, int n_in,
                              void* d_out, int out_size, void* d_ws, size_t ws_size,
                              hipStream_t stream)
{
    const float* x      = (const float*)d_in[0];  // [16,128,128,128]
    const float* style  = (const float*)d_in[1];  // [16,512]
    const float* weight = (const float*)d_in[2];  // [1,128,128,3,3]
    const float* mod_w  = (const float*)d_in[3];  // [128,512]
    const float* mod_b  = (const float*)d_in[4];  // [128]
    float* out = (float*)d_out;                   // [16,128,128,128]

    // workspace: s (8 KB) then packed bf16 weights (4.5 MB)
    float*  s     = (float*)d_ws;
    __bf16* wpack = (__bf16*)((char*)d_ws + 8192);

    style_mod_kernel<<<(B_ * CIN + 255) / 256, 256, 0, stream>>>(style, mod_w, mod_b, s);
    pack_weights_kernel<<<B_ * COUT, 128, 0, stream>>>(weight, s, wpack);

    size_t lds_bytes = (size_t)(3 * COUT * CIN * 2) + (size_t)(130 * 32 * 2);
    modconv_wmma_kernel<<<B_ * H_, 256, lds_bytes, stream>>>(x, wpack, out);
}
